// ActorCritic_11613591569095
// MI455X (gfx1250) — matrix operand order repaired and checked by reading, not compiled, not run on hardware
//
#include <hip/hip_runtime.h>

// ---------------------------------------------------------------------------
// Fused ActorCritic forward for MI455X (gfx1250, wave32, WMMA).
// bf16 hi/lo split-precision GEMM via v_wmma_f32_16x16x32_bf16 (3 wmma / K32
// chunk ~= fp32 accuracy), fused LayerNorm + SELU + second-layer GEMV.
// ---------------------------------------------------------------------------

typedef __attribute__((ext_vector_type(16))) __bf16       v16bf;
typedef __attribute__((ext_vector_type(8)))  float        v8f;
typedef __attribute__((ext_vector_type(4)))  unsigned int u32x4;

union Frag {
  v16bf v;
  u32x4 q[2];
};

__device__ __forceinline__ unsigned short f2bf(float f) {
  unsigned u = __float_as_uint(f);
  unsigned r = u + 0x7FFFu + ((u >> 16) & 1u);   // round-to-nearest-even
  return (unsigned short)(r >> 16);
}
__device__ __forceinline__ float bf2f(unsigned short h) {
  return __uint_as_float(((unsigned)h) << 16);
}

#define ASTRIDE 552   // halves per LDS A row (544 K cols + 8 pad, rows 16B aligned)
#define HSTRIDE 516   // floats per LDS hidden row (512 + 4: 8*516 % 64 == 32 -> no 2-way bank clash)
#define NCHUNK  17    // K chunks of 32: chunk0 = padded head, 1..16 = x block
#define WFRAG_HALVES ((size_t)NCHUNK * 32 * 32 * 16)   // per hi or lo array

// --------------------------- pre-pass kernels ------------------------------

__global__ void zero_flags_kernel(int* __restrict__ flags, int n) {
  int i = blockIdx.x * blockDim.x + threadIdx.x;
  if (i < n) flags[i] = 0;
}

__global__ void set_flags_kernel(const int* __restrict__ idx, int* __restrict__ flags, int n) {
  int i = blockIdx.x * blockDim.x + threadIdx.x;
  if (i < n) flags[idx[i]] = 1;   // races are benign (all write 1)
}

// Pack W1 (rows = K input dim, cols = 512 outputs) into per-lane WMMA B
// fragments, bf16 hi + lo.  Fragment (c,t): lanes 0-15 hold K=c*32+e (N=lane),
// lanes 16-31 hold K=c*32+16+e (N=lane-16), element e in [0,16).
// Chunk 0 is the zero-padded head (headK = 15 actor / 21 critic).
__global__ void pack_w1_kernel(const float* __restrict__ W, int headK,
                               unsigned short* __restrict__ whi,
                               unsigned short* __restrict__ wlo) {
  int gid = blockIdx.x * blockDim.x + threadIdx.x;     // (c*32 + t)*32 + lane
  if (gid >= NCHUNK * 32 * 32) return;
  int lane = gid & 31;
  int t    = (gid >> 5) & 31;
  int c    = gid >> 10;
  int col  = t * 16 + (lane & 15);
  int kk0  = (lane < 16) ? 0 : 16;
#pragma unroll
  for (int e = 0; e < 16; ++e) {
    int kk = kk0 + e;
    float v = 0.f;
    if (c == 0) {
      if (kk < headK) v = W[kk * 512 + col];
    } else {
      v = W[(headK + (c - 1) * 32 + kk) * 512 + col];
    }
    unsigned short hi = f2bf(v);
    whi[(size_t)gid * 16 + e] = hi;
    wlo[(size_t)gid * 16 + e] = f2bf(v - bf2f(hi));
  }
}

// ------------------------------ main kernel --------------------------------
// grid: (N/64, 2)  y==0 actor, y==1 critic.  512 threads = 16 wave32.
// Each wave owns a 32-wide column strip of the 64x512 hidden block:
// 4 M-tiles x 2 N-tiles = 8 C tiles (64 accumulator VGPRs).

__global__ __launch_bounds__(512)
void fused_actor_critic_kernel(
    const float* __restrict__ features, const int* __restrict__ flags,
    const float* __restrict__ jerr, const float* __restrict__ jdedt,
    const float* __restrict__ wts,  const float* __restrict__ acts,
    const unsigned short* __restrict__ wAhi, const unsigned short* __restrict__ wAlo,
    const unsigned short* __restrict__ wChi, const unsigned short* __restrict__ wClo,
    const float* __restrict__ b1a, const float* __restrict__ g1a, const float* __restrict__ be1a,
    const float* __restrict__ W2a, const float* __restrict__ b2a,
    const float* __restrict__ b1c, const float* __restrict__ g1c, const float* __restrict__ be1c,
    const float* __restrict__ W2c, const float* __restrict__ b2c,
    float* __restrict__ out, int N)
{
  extern __shared__ char smem[];
  unsigned short* Ahi = (unsigned short*)smem;                 // [64][ASTRIDE]
  unsigned short* Alo = Ahi + 64 * ASTRIDE;                    // [64][ASTRIDE]
  float* H    = (float*)smem;                                  // aliases A after K loop
  float* red  = (float*)(smem + (size_t)2 * 64 * ASTRIDE * 2); // [64][8][2]
  float* stat = red + 64 * 8 * 2;                              // [64][2] mu, rsigma

  const int tid  = threadIdx.x;
  const int net  = blockIdx.y;
  const int row0 = blockIdx.x * 64;

  const unsigned short* Whi = net ? wChi : wAhi;
  const unsigned short* Wlo = net ? wClo : wAlo;
  const float* b1  = net ? b1c  : b1a;
  const float* g1  = net ? g1c  : g1a;
  const float* be1 = net ? be1c : be1a;

  // ---- stage x block (cols 32..543) into LDS as bf16 hi/lo, masked by flags
  {
    int sub = tid & 7, r = tid >> 3;
    int row = row0 + r;
    bool present = flags[row] != 0;
    const float4* src = (const float4*)(features + (size_t)row * 512);
#pragma unroll 4
    for (int i = 0; i < 16; ++i) {
      int c4 = sub * 16 + i;                 // float4 index in row: 0..127
      float4 f;
      if (present) f = src[c4];
      else { f.x = 0.f; f.y = 0.f; f.z = 0.f; f.w = 0.f; }
      float vv[4] = {f.x, f.y, f.z, f.w};
      int cb = 32 + c4 * 4;
#pragma unroll
      for (int k = 0; k < 4; ++k) {
        unsigned short hi = f2bf(vv[k]);
        Ahi[r * ASTRIDE + cb + k] = hi;
        Alo[r * ASTRIDE + cb + k] = f2bf(vv[k] - bf2f(hi));
      }
    }
  }
  // ---- stage head (cols 0..31): [jerr(6)|jdedt(6)|wts(3)|(actions(6))|0...]
  if (tid < 64) {
    int r = tid, row = row0 + r;
    float h[32];
#pragma unroll
    for (int i = 0; i < 32; ++i) h[i] = 0.f;
#pragma unroll
    for (int j = 0; j < 6; ++j) h[j]     = jerr [row * 6 + j];
#pragma unroll
    for (int j = 0; j < 6; ++j) h[6 + j] = jdedt[row * 6 + j];
#pragma unroll
    for (int j = 0; j < 3; ++j) h[12 + j] = wts [row * 3 + j];
    if (net) {
#pragma unroll
      for (int j = 0; j < 6; ++j) h[15 + j] = acts[row * 6 + j];
    }
#pragma unroll
    for (int i = 0; i < 32; ++i) {
      unsigned short hi = f2bf(h[i]);
      Ahi[r * ASTRIDE + i] = hi;
      Alo[r * ASTRIDE + i] = f2bf(h[i] - bf2f(hi));
    }
  }
  __syncthreads();

  // ---- WMMA main loop over 17 K-chunks ----
  const int wave = tid >> 5, lane = tid & 31;
  const int hsel = lane >> 4;        // which half-wave
  const int lrow = lane & 15;

  v8f acc[4][2];
#pragma unroll
  for (int mt = 0; mt < 4; ++mt)
#pragma unroll
    for (int nt = 0; nt < 2; ++nt)
      acc[mt][nt] = (v8f){0.f, 0.f, 0.f, 0.f, 0.f, 0.f, 0.f, 0.f};

  for (int c = 0; c < NCHUNK; ++c) {
#pragma unroll
    for (int nt = 0; nt < 2; ++nt) {
      int t = wave * 2 + nt;
      size_t boff = ((size_t)((c * 32 + t) * 32 + lane)) * 16;   // halves
      Frag bhi, blo;
      bhi.q[0] = *(const u32x4*)(Whi + boff);
      bhi.q[1] = *(const u32x4*)(Whi + boff + 8);
      blo.q[0] = *(const u32x4*)(Wlo + boff);
      blo.q[1] = *(const u32x4*)(Wlo + boff + 8);
#pragma unroll
      for (int mt = 0; mt < 4; ++mt) {
        // A fragment: lane<16 -> row mt*16+lrow, K {c32+0..15}
        //             lane>=16 ->              K {c32+16..31}
        int ar = mt * 16 + lrow;
        int kb = c * 32 + hsel * 16;
        const unsigned short* pa = Ahi + ar * ASTRIDE + kb;
        const unsigned short* pl = Alo + ar * ASTRIDE + kb;
        Frag fahi, falo;
        fahi.q[0] = *(const u32x4*)(pa);
        fahi.q[1] = *(const u32x4*)(pa + 8);
        falo.q[0] = *(const u32x4*)(pl);
        falo.q[1] = *(const u32x4*)(pl + 8);
        // split-precision: hi*hi + hi*lo + lo*hi  (lo*lo negligible)
        acc[mt][nt] = __builtin_amdgcn_wmma_f32_16x16x32_bf16(
            false, fahi.v, false, bhi.v, (short)0, acc[mt][nt], false, false);
        acc[mt][nt] = __builtin_amdgcn_wmma_f32_16x16x32_bf16(
            false, fahi.v, false, blo.v, (short)0, acc[mt][nt], false, false);
        acc[mt][nt] = __builtin_amdgcn_wmma_f32_16x16x32_bf16(
            false, falo.v, false, bhi.v, (short)0, acc[mt][nt], false, false);
      }
    }
  }
  __syncthreads();   // A region dead -> safe to alias as hidden H

  // ---- write hidden (+ b1 bias) into LDS ----
#pragma unroll
  for (int nt = 0; nt < 2; ++nt) {
    int col = (wave * 2 + nt) * 16 + lrow;
    float bias = b1[col];
#pragma unroll
    for (int mt = 0; mt < 4; ++mt) {
#pragma unroll
      for (int j = 0; j < 8; ++j) {
        int r = mt * 16 + hsel * 8 + j;      // C layout: M = j + 8*(lane>=16)
        H[r * HSTRIDE + col] = acc[mt][nt][j] + bias;
      }
    }
  }
  __syncthreads();

  // ---- layernorm stats: 8 threads / row ----
  {
    int r = tid >> 3, sub = tid & 7;
    float s = 0.f, sq = 0.f;
    for (int i = 0; i < 64; ++i) {
      float v = H[r * HSTRIDE + sub * 64 + i];
      s += v; sq += v * v;
    }
    red[(r * 8 + sub) * 2]     = s;
    red[(r * 8 + sub) * 2 + 1] = sq;
  }
  __syncthreads();
  if (tid < 64) {
    float s = 0.f, sq = 0.f;
#pragma unroll
    for (int p = 0; p < 8; ++p) {
      s  += red[(tid * 8 + p) * 2];
      sq += red[(tid * 8 + p) * 2 + 1];
    }
    float mu  = s * (1.f / 512.f);
    float var = sq * (1.f / 512.f) - mu * mu;
    stat[tid * 2]     = mu;
    stat[tid * 2 + 1] = rsqrtf(var + 1e-5f);
  }
  __syncthreads();

  // ---- LN + SELU in place ----
  {
    const float SC = 1.0507009873554805f, AL = 1.6732632423543772f;
    int r = tid >> 3, sub = tid & 7;
    float mu = stat[r * 2], rs = stat[r * 2 + 1];
    for (int i = 0; i < 64; ++i) {
      int col = sub * 64 + i;
      float v = H[r * HSTRIDE + col];
      v = (v - mu) * rs * g1[col] + be1[col];
      v = (v > 0.f) ? SC * v : SC * AL * (expf(v) - 1.f);
      H[r * HSTRIDE + col] = v;
    }
  }
  __syncthreads();

  // ---- layer 2 GEMV + output ----
  if (net == 0) {
    if (tid < 384) {                       // 64 rows x 6 outputs
      int r = tid / 6, j = tid % 6;
      float s = 0.f;
      for (int c = 0; c < 512; ++c) s += H[r * HSTRIDE + c] * W2a[c * 6 + j];
      out[(size_t)(row0 + r) * 6 + j] = tanhf(s + b2a[j]);
    }
  } else {
    if (tid < 64) {                        // 64 rows x 1 output
      int r = tid;
      float s = 0.f;
      for (int c = 0; c < 512; ++c) s += H[r * HSTRIDE + c] * W2c[c];
      out[(size_t)N * 6 + row0 + r] = s + b2c[0];
    }
  }
}

// ------------------------------ launcher -----------------------------------

extern "C" void kernel_launch(void* const* d_in, const int* in_sizes, int n_in,
                              void* d_out, int out_size, void* d_ws, size_t ws_size,
                              hipStream_t stream) {
  const float* features = (const float*)d_in[0];
  const int*   idx      = (const int*)  d_in[1];
  const float* jerr     = (const float*)d_in[2];
  const float* jdedt    = (const float*)d_in[3];
  const float* wts      = (const float*)d_in[4];
  const float* acts     = (const float*)d_in[5];
  const float* W1a      = (const float*)d_in[6];
  const float* b1a      = (const float*)d_in[7];
  const float* g1a      = (const float*)d_in[8];
  const float* be1a     = (const float*)d_in[9];
  const float* W2a      = (const float*)d_in[10];
  const float* b2a      = (const float*)d_in[11];
  const float* W1c      = (const float*)d_in[12];
  const float* b1c      = (const float*)d_in[13];
  const float* g1c      = (const float*)d_in[14];
  const float* be1c     = (const float*)d_in[15];
  const float* W2c      = (const float*)d_in[16];
  const float* b2c      = (const float*)d_in[17];
  float* out = (float*)d_out;

  const int N = in_sizes[1];               // idx length

  // workspace: flags[N] + 4 packed weight arrays (bf16 hi/lo, actor/critic)
  char* p = (char*)d_ws;
  int* flags = (int*)p;              p += (size_t)N * sizeof(int);
  unsigned short* wAhi = (unsigned short*)p; p += WFRAG_HALVES * 2;
  unsigned short* wAlo = (unsigned short*)p; p += WFRAG_HALVES * 2;
  unsigned short* wChi = (unsigned short*)p; p += WFRAG_HALVES * 2;
  unsigned short* wClo = (unsigned short*)p; p += WFRAG_HALVES * 2;

  zero_flags_kernel<<<(N + 255) / 256, 256, 0, stream>>>(flags, N);
  set_flags_kernel <<<(N + 255) / 256, 256, 0, stream>>>(idx, flags, N);

  const int packThreads = NCHUNK * 32 * 32;
  pack_w1_kernel<<<(packThreads + 255) / 256, 256, 0, stream>>>(W1a, 15, wAhi, wAlo);
  pack_w1_kernel<<<(packThreads + 255) / 256, 256, 0, stream>>>(W1c, 21, wChi, wClo);

  size_t shmem = (size_t)2 * 64 * ASTRIDE * sizeof(unsigned short)   // A hi+lo (aliased by H)
               + (size_t)(64 * 8 * 2 + 64 * 2) * sizeof(float);      // reduction scratch
  (void)hipFuncSetAttribute((const void*)fused_actor_critic_kernel,
                            hipFuncAttributeMaxDynamicSharedMemorySize, (int)shmem);

  dim3 grid(N / 64, 2);
  fused_actor_critic_kernel<<<grid, 512, shmem, stream>>>(
      features, flags, jerr, jdedt, wts, acts,
      wAhi, wAlo, wChi, wClo,
      b1a, g1a, be1a, W2a, b2a,
      b1c, g1c, be1c, W2c, b2c,
      out, N);
}
